// EventPixelFF_76295799046508
// MI455X (gfx1250) — matrix operand order repaired and checked by reading, not compiled, not run on hardware
//
#include <hip/hip_runtime.h>

// ---------------------------------------------------------------------------
// Problem constants (match reference)
// ---------------------------------------------------------------------------
#define WDIM 320
#define HDIM 180
#define BDIM 2
#define TDIM 20
#define LLVL 8
#define TBLSZ (1 << 19)
#define NEV 262144
#define NPIX (BDIM * WDIM * HDIM) /* 115200 = 900*128 */

typedef __attribute__((ext_vector_type(16))) __bf16 v16bf;
typedef __attribute__((ext_vector_type(8))) float v8f;
typedef __attribute__((ext_vector_type(4))) unsigned int u32x4;
typedef __attribute__((ext_vector_type(8))) int i32x8;
typedef __attribute__((ext_vector_type(4))) float f32x4;

union ABx {
    u32x4 u[2];
    v16bf v;
};

static __device__ inline unsigned short f2bf(float f) {
    unsigned int u = __builtin_bit_cast(unsigned int, f);
    unsigned int r = u + 0x7FFFu + ((u >> 16) & 1u); // round-to-nearest-even
    return (unsigned short)(r >> 16);
}

#define WMMA_BF16(a, b, c) \
    __builtin_amdgcn_wmma_f32_16x16x32_bf16(false, (a), false, (b), (short)0, (c), false, false)

// CDNA5 async global->LDS copy (ASYNCcnt-tracked). lds_off = low 32 bits of a
// generic pointer to LDS (== byte offset within the workgroup LDS allocation).
static __device__ inline void async_b128(unsigned lds_off, unsigned long long gaddr) {
    asm volatile("global_load_async_to_lds_b128 %0, %1, off" ::"v"(lds_off), "v"(gaddr)
                 : "memory");
}
#define WAIT_ASYNC3() asm volatile("s_wait_asynccnt 0x3" ::: "memory")
#define WAIT_ASYNC0() asm volatile("s_wait_asynccnt 0x0" ::: "memory")

// A fragment (16x32 bf16, MxK) from a row-major [rows][rowStrideU4*8 bf16] tile.
// Contiguous k order (same as B packing): lanes 0-15 -> M=lane, K=0..7 (u[0]) and 8..15 (u[1]);
//             lanes 16-31 -> M=lane-16, K=16..23 (u[0]) and 24..31 (u[1]).
static __device__ inline v16bf load_afrag(const u32x4* base, int rowStrideU4, int row, int s,
                                          int lane) {
    ABx t;
    const u32x4* rp = base + (size_t)row * rowStrideU4 + s * 4;
    const int hi = lane >> 4;
    t.u[0] = rp[2 * hi];
    t.u[1] = rp[2 * hi + 1];
    return t.v;
}

// B fragment (32x16 bf16, KxN) from pre-packed fragment layout
// wf[s][g][lane][16 bf16], lanes 0-15: K=0..15 at N=g*16+lane,
// lanes 16-31: K=16..31 at N=g*16+(lane-16).
static __device__ inline v16bf load_bfrag(const u32x4* wf, int G, int s, int g, int lane) {
    ABx t;
    const u32x4* q = wf + (((size_t)s * G + g) * 32 + lane) * 2;
    t.u[0] = q[0];
    t.u[1] = q[1];
    return t.v;
}

// Same fragment but from an LDS slab holding one K-step of 4 groups.
static __device__ inline v16bf load_bfrag_lds(const u32x4* sb, int g, int lane) {
    ABx t;
    const u32x4* q = sb + (g * 32 + lane) * 2;
    t.u[0] = q[0];
    t.u[1] = q[1];
    return t.v;
}

// ---------------------------------------------------------------------------
// Utility kernels
// ---------------------------------------------------------------------------
__global__ void zero_f32x4(float* p, int n4) {
    int i = blockIdx.x * blockDim.x + threadIdx.x;
    if (i < n4) {
        f32x4 z = {0.f, 0.f, 0.f, 0.f};
        ((f32x4*)p)[i] = z;
    }
}

__global__ void cvt_f32_to_bf16(const float* __restrict__ in, unsigned short* __restrict__ out,
                                int n) {
    int i = blockIdx.x * blockDim.x + threadIdx.x;
    if (i < n) out[i] = f2bf(in[i]);
}

// Re-pack weights (conv OIHW f32, or MLP (K,N) f32) into the WMMA B-fragment
// layout, flat index t == fragment flat index.
__global__ void reorder_w(const float* __restrict__ w, unsigned short* __restrict__ dst, int CI,
                          int CO, int CO_real, int is_conv, int total) {
    int t = blockIdx.x * blockDim.x + threadIdx.x;
    if (t >= total) return;
    const int e = t & 15;
    const int lane = (t >> 4) & 31;
    const int rest = t >> 9;
    const int G = CO >> 4;
    const int g = rest % G;
    const int s = rest / G;
    const int k = 32 * s + ((lane >> 4) << 4) + e;
    const int n = g * 16 + (lane & 15);
    float v = 0.f;
    if (n < CO_real) {
        if (is_conv) {
            const int tap = k / CI, ci = k - tap * CI;
            const int dx = tap / 3, dy = tap % 3;
            v = w[(((size_t)n * CI + ci) * 3 + dx) * 3 + dy];
        } else {
            v = w[(size_t)k * CO_real + n];
        }
    }
    dst[t] = f2bf(v);
}

// ---------------------------------------------------------------------------
// Hash encode + scatter-add (bandwidth bound: gathers + f32 atomics)
// ---------------------------------------------------------------------------
__global__ void hash_scatter(const float* __restrict__ coords, const int* __restrict__ counts,
                             const float* __restrict__ tables, float* __restrict__ grid) {
    int i = blockIdx.x * blockDim.x + threadIdx.x;
    if (i >= NEV) return;
    const float c0 = coords[3 * i + 0];
    const float c1 = coords[3 * i + 1];
    const float c2 = coords[3 * i + 2];
    int b = 0, acc = 0;
    for (int j = 0; j < BDIM; ++j) {
        if (i >= acc) b = j;
        acc += counts[j];
    }
    int cx = (int)fminf(fmaxf(roundf(c0 * (float)WDIM), 0.f), (float)(WDIM - 1));
    int cy = (int)fminf(fmaxf(roundf(c1 * (float)HDIM), 0.f), (float)(HDIM - 1));
    float* gp = grid + (((size_t)b * WDIM + cx) * HDIM + cy) * 32;

    const float gpl0 = __logf(320.f / 16.f) / 7.f;
    const float gpl1 = __logf(180.f / 16.f) / 7.f;
    const float gpl2 = __logf(50.f / 4.f) / 7.f;

    for (int l = 0; l < LLVL; ++l) {
        const float r0 = (float)(int)(16.f * __expf(gpl0 * (float)l) + 1e-3f);
        const float r1 = (float)(int)(16.f * __expf(gpl1 * (float)l) + 1e-3f);
        const float r2 = (float)(int)(4.f * __expf(gpl2 * (float)l) + 1e-3f);
        const float s0 = c0 * r0, s1 = c1 * r1, s2 = c2 * r2;
        const float f0 = floorf(s0), f1 = floorf(s1), f2 = floorf(s2);
        const float fr0 = s0 - f0, fr1 = s1 - f1, fr2 = s2 - f2;
        const unsigned int b0 = (unsigned int)f0;
        const unsigned int b1 = (unsigned int)f1;
        const unsigned int b2 = (unsigned int)f2;
        float feat0 = 0.f, feat1 = 0.f, feat2 = 0.f, feat3 = 0.f;
#pragma unroll
        for (int c = 0; c < 8; ++c) {
            const unsigned int o0 = c & 1, o1 = (c >> 1) & 1, o2 = (c >> 2) & 1;
            const unsigned int h =
                (((b0 + o0) * 1u) ^ ((b1 + o1) * 2654435761u) ^ ((b2 + o2) * 805459861u)) &
                (TBLSZ - 1);
            const f32x4 f = *(const f32x4*)(tables + ((size_t)l * TBLSZ + h) * 4);
            const float wsel = (o0 ? fr0 : 1.f - fr0) * (o1 ? fr1 : 1.f - fr1) *
                               (o2 ? fr2 : 1.f - fr2);
            feat0 += wsel * f.x;
            feat1 += wsel * f.y;
            feat2 += wsel * f.z;
            feat3 += wsel * f.w;
        }
        atomicAdd(gp + l * 4 + 0, feat0);
        atomicAdd(gp + l * 4 + 1, feat1);
        atomicAdd(gp + l * 4 + 2, feat2);
        atomicAdd(gp + l * 4 + 3, feat3);
    }
}

// ---------------------------------------------------------------------------
// Implicit-GEMM 3x3 conv, NHWC bf16, WMMA bf16 f32-accumulate, bias+ReLU.
// Block: 256 threads (8 waves) -> 128-pixel x 64-outchannel tile.
// Per K-step (one tap x 32 input channels): A tile (128x32) and B slab
// (32x64 weights) are staged LDS-side with async global->LDS copies,
// double-buffered so staging overlaps the WMMAs (ASYNCcnt pipeline).
// Epilogue assembles the 128x64 bf16 output tile in LDS and streams it out
// with one Tensor-DMA store (TENSORcnt-tracked).
// ---------------------------------------------------------------------------
template <int CI, int CO>
__global__ __launch_bounds__(256) void conv_wmma(const unsigned short* __restrict__ in,
                                                 const unsigned short* __restrict__ wfrag,
                                                 const float* __restrict__ bias,
                                                 unsigned short* __restrict__ out) {
    constexpr int NSTEP = (9 * CI) / 32;
    constexpr int G = CO / 16;
    __shared__ u32x4 sA[2][512];          // 2 x (128 px x 32 bf16) = 2 x 8KB
    __shared__ u32x4 sB[2][256];          // 2 x (32 K x 64 N bf16) = 2 x 4KB
    __shared__ unsigned short sO[128 * 64]; // output tile, 16KB

    const int tid = threadIdx.x;
    const int lane = tid & 31;
    const int wave = tid >> 5;
    const int p0 = blockIdx.x * 128;
    const int g0 = blockIdx.y * 4; // 4 groups of 16 out channels

    // staging role: thread -> (pixel row, 32B half)
    const int ip = tid >> 1;
    const int half = tid & 1;
    const int p = p0 + ip;
    const int b = p / (WDIM * HDIM);
    const int rem = p - b * (WDIM * HDIM);
    const int x = rem / HDIM;
    const int y = rem - x * HDIM;

    const u32x4* wfU4 = (const u32x4*)wfrag;

    // issue one K-step's async copies into buffer `buf`
    auto stage = [&](int s, int buf) {
        const int kg = 32 * s;
        const int tap = kg / CI;
        const int ci0 = kg - tap * CI;
        const int xs = x + tap / 3 - 1;
        const int ys = y + tap % 3 - 1;
        u32x4* adst = &sA[buf][ip * 4 + half * 2];
        if (xs >= 0 && xs < WDIM && ys >= 0 && ys < HDIM) {
            const unsigned long long src = (unsigned long long)(size_t)(
                in + (((size_t)b * WDIM + xs) * HDIM + ys) * CI + ci0 + half * 16);
            const unsigned aoff = (unsigned)(size_t)adst;
            async_b128(aoff, src);
            async_b128(aoff + 16, src + 16);
        } else {
            u32x4 z = {0u, 0u, 0u, 0u};
            adst[0] = z;
            adst[1] = z;
        }
        const unsigned long long bsrc =
            (unsigned long long)(size_t)(wfU4 + ((size_t)s * G + g0) * 64 + tid);
        async_b128((unsigned)(size_t)&sB[buf][tid], bsrc);
    };

    v8f acc[4] = {};
    stage(0, 0);
    for (int s = 0; s < NSTEP; ++s) {
        const int cur = s & 1;
        if (s + 1 < NSTEP) {
            stage(s + 1, 1 - cur);
            WAIT_ASYNC3(); // K-step s fully landed; s+1 still in flight
        } else {
            WAIT_ASYNC0();
        }
        __syncthreads();

        if (lane == 0 && s + 2 < NSTEP) // prefetch K-step s+2 weights toward L2
            __builtin_prefetch((const void*)(wfU4 + ((size_t)(s + 2) * G + g0) * 64), 0, 2);

        const v16bf a = load_afrag(&sA[cur][0], 4, wave * 16 + (lane & 15), 0, lane);
#pragma unroll
        for (int g = 0; g < 4; ++g) {
            const v16bf bm = load_bfrag_lds(&sB[cur][0], g, lane);
            acc[g] = WMMA_BF16(a, bm, acc[g]);
        }
        __syncthreads(); // all waves done with buffer `cur` before it refills
    }

    // epilogue: bias + ReLU -> LDS tile (C layout: lane&15 -> N, (lane>>4)*8+r -> M)
    const int col = lane & 15;
    const int mhi = (lane >> 4) * 8;
#pragma unroll
    for (int g = 0; g < 4; ++g) {
        const float bb = bias[(g0 + g) * 16 + col];
#pragma unroll
        for (int r = 0; r < 8; ++r) {
            float v = acc[g][r] + bb;
            v = v > 0.f ? v : 0.f;
            sO[(wave * 16 + mhi + r) * 64 + g * 16 + col] = f2bf(v);
        }
    }
    __syncthreads();

    // One TDM store: 2D tile, 128 rows x 64 bf16, row stride CO elements.
    if (tid < 32) {
        const unsigned long long ga =
            (unsigned long long)(size_t)(out + (size_t)p0 * CO + g0 * 16);
        u32x4 d0;
        d0.x = 1u; // count=1, user descriptor
        d0.y = (unsigned)(size_t)&sO[0];
        d0.z = (unsigned)(ga & 0xFFFFFFFFu);
        d0.w = (unsigned)((ga >> 32) & 0x1FFFFFFu) | (2u << 30); // type=2 (image)
        i32x8 d1;
        d1[0] = 1 << 16;              // data_size = 2 bytes
        d1[1] = 0;                    // tensor_dim0[15:0] (dim0 = 1<<20)
        d1[2] = (1 << 20) >> 16;      // tensor_dim0[31:16] | tensor_dim1[15:0]
        d1[3] = ((1 << 20) >> 16) | (64 << 16); // tensor_dim1[31:16] | tile_dim0=64
        d1[4] = 128;                  // tile_dim1=128, tile_dim2=0
        d1[5] = CO;                   // tensor_dim0_stride low 32 (elements)
        d1[6] = 0;
        d1[7] = 0;
        asm volatile("tensor_store_from_lds %0, %1" ::"s"(d0), "s"(d1) : "memory");
    }
#if __has_builtin(__builtin_amdgcn_s_wait_tensorcnt)
    __builtin_amdgcn_s_wait_tensorcnt(0);
#else
    asm volatile("s_wait_tensorcnt 0x0" ::: "memory");
#endif
}

// ---------------------------------------------------------------------------
// Fused per-pixel MLP 128->128->64->32->20(+pad), all WMMA; intermediates
// round-trip through a per-wave LDS slab (C-layout -> A-fragment relayout).
// Block: 256 threads = 8 waves, each wave owns 16 pixels.
// ---------------------------------------------------------------------------
__global__ __launch_bounds__(256) void mlp_wmma(
    const unsigned short* __restrict__ x, const unsigned short* __restrict__ w0,
    const float* __restrict__ b0, const unsigned short* __restrict__ w1,
    const float* __restrict__ b1, const unsigned short* __restrict__ w2,
    const float* __restrict__ b2, const unsigned short* __restrict__ w3,
    const float* __restrict__ b3, float* __restrict__ out) {
    __shared__ u32x4 hbuf4[8][256]; // 4KB per wave (16 x 128 bf16)
    const int tid = threadIdx.x;
    const int lane = tid & 31;
    const int wave = tid >> 5;
    const int pbase = blockIdx.x * 128 + wave * 16;
    unsigned short* h = (unsigned short*)&hbuf4[wave][0];
    u32x4* hU4 = &hbuf4[wave][0];
    const int col = lane & 15;
    const int mhi = (lane >> 4) * 8;
    const int row = lane & 15;
    const u32x4* xU4 = (const u32x4*)x;

    // layer 0: 128 -> 128 (A read straight from global NHWC activations)
    v8f a0[8] = {};
    for (int s = 0; s < 4; ++s) {
        const v16bf a = load_afrag(xU4, 16, pbase + row, s, lane);
#pragma unroll
        for (int g = 0; g < 8; ++g)
            a0[g] = WMMA_BF16(a, load_bfrag((const u32x4*)w0, 8, s, g, lane), a0[g]);
    }
#pragma unroll
    for (int g = 0; g < 8; ++g) {
        const float bb = b0[g * 16 + col];
#pragma unroll
        for (int r = 0; r < 8; ++r) {
            float v = a0[g][r] + bb;
            v = v > 0.f ? v : 0.f;
            h[(mhi + r) * 128 + g * 16 + col] = f2bf(v);
        }
    }

    // layer 1: 128 -> 64
    v8f a1[4] = {};
    for (int s = 0; s < 4; ++s) {
        const v16bf a = load_afrag(hU4, 16, row, s, lane);
#pragma unroll
        for (int g = 0; g < 4; ++g)
            a1[g] = WMMA_BF16(a, load_bfrag((const u32x4*)w1, 4, s, g, lane), a1[g]);
    }
#pragma unroll
    for (int g = 0; g < 4; ++g) {
        const float bb = b1[g * 16 + col];
#pragma unroll
        for (int r = 0; r < 8; ++r) {
            float v = a1[g][r] + bb;
            v = v > 0.f ? v : 0.f;
            h[(mhi + r) * 64 + g * 16 + col] = f2bf(v);
        }
    }

    // layer 2: 64 -> 32
    v8f a2[2] = {};
    for (int s = 0; s < 2; ++s) {
        const v16bf a = load_afrag(hU4, 8, row, s, lane);
#pragma unroll
        for (int g = 0; g < 2; ++g)
            a2[g] = WMMA_BF16(a, load_bfrag((const u32x4*)w2, 2, s, g, lane), a2[g]);
    }
#pragma unroll
    for (int g = 0; g < 2; ++g) {
        const float bb = b2[g * 16 + col];
#pragma unroll
        for (int r = 0; r < 8; ++r) {
            float v = a2[g][r] + bb;
            v = v > 0.f ? v : 0.f;
            h[(mhi + r) * 32 + g * 16 + col] = f2bf(v);
        }
    }

    // layer 3: 32 -> 20 (N padded to 32), f32 output, no ReLU
    v8f a3[2] = {};
    {
        const v16bf a = load_afrag(hU4, 4, row, 0, lane);
#pragma unroll
        for (int g = 0; g < 2; ++g)
            a3[g] = WMMA_BF16(a, load_bfrag((const u32x4*)w3, 2, 0, g, lane), a3[g]);
    }
#pragma unroll
    for (int g = 0; g < 2; ++g) {
        const int n = g * 16 + col;
        if (n < TDIM) {
            const float bb = b3[n];
#pragma unroll
            for (int r = 0; r < 8; ++r) {
                const int pm = pbase + mhi + r;
                out[(size_t)pm * TDIM + n] = a3[g][r] + bb;
            }
        }
    }
}

// ---------------------------------------------------------------------------
// Host orchestration
// ---------------------------------------------------------------------------
extern "C" void kernel_launch(void* const* d_in, const int* in_sizes, int n_in, void* d_out,
                              int out_size, void* d_ws, size_t ws_size, hipStream_t stream) {
    (void)in_sizes;
    (void)n_in;
    (void)out_size;
    (void)ws_size;

    const float* coords = (const float*)d_in[0];
    const int* counts = (const int*)d_in[1];
    const float* tables = (const float*)d_in[2];
    const float* c1w = (const float*)d_in[3];
    const float* c1b = (const float*)d_in[4];
    const float* c2w = (const float*)d_in[5];
    const float* c2b = (const float*)d_in[6];
    const float* c3w = (const float*)d_in[7];
    const float* c3b = (const float*)d_in[8];
    const float* c4w = (const float*)d_in[9];
    const float* c4b = (const float*)d_in[10];
    const float* m0w = (const float*)d_in[11];
    const float* m0b = (const float*)d_in[12];
    const float* m1w = (const float*)d_in[13];
    const float* m1b = (const float*)d_in[14];
    const float* m2w = (const float*)d_in[15];
    const float* m2b = (const float*)d_in[16];
    const float* m3w = (const float*)d_in[17];
    const float* m3b = (const float*)d_in[18];

    char* ws = (char*)d_ws;
    size_t cur = 0;
    auto walloc = [&](size_t bytes) -> char* {
        char* q = ws + cur;
        cur = (cur + bytes + 255) & ~(size_t)255;
        return q;
    };
    float* grid = (float*)walloc((size_t)NPIX * 32 * sizeof(float));
    unsigned short* actA = (unsigned short*)walloc((size_t)NPIX * 256 * 2);
    unsigned short* actB = (unsigned short*)walloc((size_t)NPIX * 256 * 2);
    unsigned short* wf1 = (unsigned short*)walloc((size_t)9 * 32 * 256 * 2);  // K=288,N=256
    unsigned short* wf2 = (unsigned short*)walloc((size_t)9 * 256 * 256 * 2); // K=2304,N=256
    unsigned short* wf3 = (unsigned short*)walloc((size_t)9 * 256 * 256 * 2);
    unsigned short* wf4 = (unsigned short*)walloc((size_t)9 * 256 * 128 * 2); // K=2304,N=128
    unsigned short* wm0 = (unsigned short*)walloc((size_t)128 * 128 * 2);
    unsigned short* wm1 = (unsigned short*)walloc((size_t)128 * 64 * 2);
    unsigned short* wm2 = (unsigned short*)walloc((size_t)64 * 32 * 2);
    unsigned short* wm3 = (unsigned short*)walloc((size_t)32 * 32 * 2); // N padded 20->32

    // 1) zero the dense scatter grid
    const int n4 = NPIX * 32 / 4;
    zero_f32x4<<<(n4 + 255) / 256, 256, 0, stream>>>(grid, n4);

    // 2) hash-encode events + scatter-add
    hash_scatter<<<NEV / 256, 256, 0, stream>>>(coords, counts, tables, grid);

    // 3) f32 grid -> bf16 NHWC activations
    const int ne0 = NPIX * 32;
    cvt_f32_to_bf16<<<(ne0 + 255) / 256, 256, 0, stream>>>(grid, actA, ne0);

    // 4) pre-pack all weights into WMMA B-fragment layout
    auto launch_reorder = [&](const float* w, unsigned short* dst, int CI, int CO, int CO_real,
                              int is_conv, int nstep) {
        const int total = nstep * (CO / 16) * 512;
        reorder_w<<<(total + 255) / 256, 256, 0, stream>>>(w, dst, CI, CO, CO_real, is_conv,
                                                           total);
    };
    launch_reorder(c1w, wf1, 32, 256, 256, 1, 9);
    launch_reorder(c2w, wf2, 256, 256, 256, 1, 72);
    launch_reorder(c3w, wf3, 256, 256, 256, 1, 72);
    launch_reorder(c4w, wf4, 256, 128, 128, 1, 72);
    launch_reorder(m0w, wm0, 128, 128, 128, 0, 4);
    launch_reorder(m1w, wm1, 128, 64, 64, 0, 4);
    launch_reorder(m2w, wm2, 64, 32, 32, 0, 2);
    launch_reorder(m3w, wm3, 32, 32, 20, 0, 1);

    // 5) conv stack (implicit GEMM, WMMA bf16, async-LDS pipeline + TDM stores)
    dim3 blk(256);
    dim3 gFull(NPIX / 128, 4); // 64 out channels per block, CO=256
    dim3 gHalf(NPIX / 128, 2); // CO=128
    conv_wmma<32, 256><<<gFull, blk, 0, stream>>>(actA, wf1, c1b, actB);
    conv_wmma<256, 256><<<gFull, blk, 0, stream>>>(actB, wf2, c2b, actA);
    conv_wmma<256, 256><<<gFull, blk, 0, stream>>>(actA, wf3, c3b, actB);
    conv_wmma<256, 128><<<gHalf, blk, 0, stream>>>(actB, wf4, c4b, actA);

    // 6) fused MLP -> logits (B,W,H,T) f32
    mlp_wmma<<<NPIX / 128, blk, 0, stream>>>(actA, wm0, m0b, wm1, m1b, wm2, m2b, wm3, m3b,
                                             (float*)d_out);
}
